// MultiHeadAttentionClassical_65481071408476
// MI455X (gfx1250) — matrix operand order repaired and checked by reading, not compiled, not run on hardware
//
#include <hip/hip_runtime.h>
#include <hip/hip_bf16.h>

// ---------------------------------------------------------------------------
// MHA forward for B=2, S=2048, D=1024, H=16, d_k=64 on gfx1250 (MI455X).
// bf16 WMMA (f32 accum), flash attention, block-cooperative async-to-LDS
// staging, and sched_group_barrier-shaped inner loops so DS tile reads run
// several tiles ahead of the WMMAs (instead of s_wait_dscnt 0x0 stalls).
// ---------------------------------------------------------------------------

typedef __bf16 bf16;
typedef __attribute__((ext_vector_type(16))) __bf16 v16bf;
typedef __attribute__((ext_vector_type(8)))  __bf16 v8bf;
typedef __attribute__((ext_vector_type(4)))  __bf16 v4bf;
typedef __attribute__((ext_vector_type(8)))  float  v8f;
typedef __attribute__((ext_vector_type(4)))  float  v4f;

#define NB   2
#define NS   2048
#define ND   1024
#define NH   16
#define NDK  64
#define NM   (NB * NS)

// scheduler shaping: mask 0x100 = DS read, 0x008 = MFMA/WMMA
#if __has_builtin(__builtin_amdgcn_sched_group_barrier)
#define SGB(mask, size) __builtin_amdgcn_sched_group_barrier(mask, size, 0)
#else
#define SGB(mask, size)
#endif

__device__ __forceinline__ v16bf load16(const bf16* p0, const bf16* p1) {
    v16bf r;
    *(v8bf*)&r       = *(const v8bf*)p0;
    *((v8bf*)&r + 1) = *(const v8bf*)p1;
    return r;
}

__device__ __forceinline__ v8f wmma_bf16(v16bf a, v16bf b, v8f c) {
    return __builtin_amdgcn_wmma_f32_16x16x32_bf16(
        false, a, false, b, (short)0, c, false, false);
}

// CDNA5 async copy global -> LDS (ISA 15.18.3 op 98), tracked by ASYNCcnt.
__device__ __forceinline__ void async_ld_b128(bf16* lds_dst, const bf16* g) {
    uint32_t loff = (uint32_t)(uintptr_t)lds_dst;
    uint64_t ga   = (uint64_t)(uintptr_t)g;
    asm volatile("global_load_async_to_lds_b128 %0, %1, off"
                 :: "v"(loff), "v"(ga) : "memory");
}

__device__ __forceinline__ float rmax16(float v) {
    v = fmaxf(v, __shfl_xor(v, 1));
    v = fmaxf(v, __shfl_xor(v, 2));
    v = fmaxf(v, __shfl_xor(v, 4));
    v = fmaxf(v, __shfl_xor(v, 8));
    return v;
}
__device__ __forceinline__ float rsum16(float v) {
    v += __shfl_xor(v, 1);
    v += __shfl_xor(v, 2);
    v += __shfl_xor(v, 4);
    v += __shfl_xor(v, 8);
    return v;
}

// ---------------------------------------------------------------------------
// Kernel 1: fp32 -> bf16 conversion.
// ---------------------------------------------------------------------------
__global__ __launch_bounds__(256) void cvt_bf16(const float* __restrict__ src,
                                                bf16* __restrict__ dst, int n) {
    int i = (blockIdx.x * 256 + threadIdx.x) * 4;
    if (i >= n) return;
    v4f f = *(const v4f*)(src + i);
    v4bf o;
    o[0] = (bf16)f[0]; o[1] = (bf16)f[1]; o[2] = (bf16)f[2]; o[3] = (bf16)f[3];
    *(v4bf*)(dst + i) = o;
}

// ---------------------------------------------------------------------------
// Kernel 2: fused QKV projection.
// Block = 8 waves; block shares an async-staged 64-column panel of Wq/Wk/Wv.
// Wave computes 16x64 of Q, K and V (12 accumulators).
// ---------------------------------------------------------------------------
#define BROW 40                       // padded LDS row stride (elements)
__global__ __launch_bounds__(256) void qkv_gemm(
    const bf16* __restrict__ xb,
    const bf16* __restrict__ wq, const bf16* __restrict__ wk,
    const bf16* __restrict__ wv,
    bf16* __restrict__ Qb, bf16* __restrict__ Kb, bf16* __restrict__ Vtb) {

    __shared__ bf16 Bq[2][64 * BROW];
    __shared__ bf16 Bk[2][64 * BROW];
    __shared__ bf16 Bv[2][64 * BROW];

    const int tid  = threadIdx.x;
    const int lane = tid & 31;
    const int w    = tid >> 5;
    const int colBlock = blockIdx.x & 15;     // 64-col panel (== head h)
    const int rowGroup = blockIdx.x >> 4;     // 0..31
    const int rowTile  = rowGroup * 8 + w;    // 0..255
    const int half = lane >> 4;
    const int l16  = lane & 15;

    const int srow = tid >> 2;
    const int scol = (tid & 3) * 8;
    const bf16* qsrc = wq + (size_t)(colBlock * 64 + srow) * ND + scol;
    const bf16* ksrc = wk + (size_t)(colBlock * 64 + srow) * ND + scol;
    const bf16* vsrc = wv + (size_t)(colBlock * 64 + srow) * ND + scol;

    const bf16* arow = xb + (size_t)(rowTile * 16 + l16) * ND;
    const int a0 = half * 16, a1 = half * 16 + 8;

    // acc[i]: weight (i%3: q,k,v), column tile nt = i/3
    v8f acc[12] = {};

    async_ld_b128(&Bq[0][srow * BROW + scol], qsrc);
    async_ld_b128(&Bk[0][srow * BROW + scol], ksrc);
    async_ld_b128(&Bv[0][srow * BROW + scol], vsrc);
    v16bf a_c = load16(arow + a0, arow + a1);

    for (int kk = 0; kk < ND / 32; ++kk) {
        if (kk + 1 < ND / 32) {
            const int nb = (kk + 1) & 1, ko = (kk + 1) * 32;
            async_ld_b128(&Bq[nb][srow * BROW + scol], qsrc + ko);
            async_ld_b128(&Bk[nb][srow * BROW + scol], ksrc + ko);
            async_ld_b128(&Bv[nb][srow * BROW + scol], vsrc + ko);
            asm volatile("s_wait_asynccnt 0x3" ::: "memory");
        } else {
            asm volatile("s_wait_asynccnt 0x0" ::: "memory");
        }
        __syncthreads();   // panel kk visible to all waves

        const int kn = (kk + 1 < ND / 32 ? kk + 1 : kk) * 32;
        v16bf a_n = load16(arow + kn + a0, arow + kn + a1);

        const int buf = kk & 1;
        const bf16* bases[3] = { &Bq[buf][0], &Bk[buf][0], &Bv[buf][0] };

        v16bf bt[12];
#pragma unroll
        for (int i = 0; i < 12; ++i) {
            const bf16* p = bases[i % 3] + ((i / 3) * 16 + l16) * BROW + half * 16;
            bt[i] = load16(p, p + 8);
        }
#pragma unroll
        for (int i = 0; i < 12; ++i)
            acc[i] = wmma_bf16(a_c, bt[i], acc[i]);

        // shape: keep 4 tiles (8 DS reads) in flight ahead of each WMMA
        SGB(0x100, 8);                 // tiles 0..3 loads
#pragma unroll
        for (int i = 0; i < 8; ++i) {
            SGB(0x008, 1);             // WMMA i
            SGB(0x100, 2);             // tile i+4 loads
        }
        SGB(0x008, 4);                 // WMMAs 8..11

        a_c = a_n;
        __syncthreads();   // everyone done reading buf before it is reused
    }

#pragma unroll
    for (int r = 0; r < 8; ++r) {
        const int m = rowTile * 16 + r + half * 8;
        const int b = m >> 11, s = m & (NS - 1);
        const int bh = b * NH + colBlock;
#pragma unroll
        for (int nt = 0; nt < 4; ++nt) {
            const int dd = nt * 16 + l16;
            Qb[((size_t)bh * NS + s) * NDK + dd]  = (bf16)acc[nt * 3 + 0][r];
            Kb[((size_t)bh * NS + s) * NDK + dd]  = (bf16)acc[nt * 3 + 1][r];
            Vtb[((size_t)bh * NDK + dd) * NS + s] = (bf16)acc[nt * 3 + 2][r];
        }
    }
}

// ---------------------------------------------------------------------------
// Kernel 3: flash attention. Block = 8 waves on the SAME (b,h); K/V blocks
// async-staged into LDS once per 32-key iteration for all waves.
// ---------------------------------------------------------------------------
#define KROW 72                       // padded K row stride (64+8 elements)
#define VROW 40                       // padded V row stride (32+8 elements)
__global__ __launch_bounds__(256) void flash_attn(
    const bf16* __restrict__ Qb, const bf16* __restrict__ Kb,
    const bf16* __restrict__ Vtb, bf16* __restrict__ Ab) {

    __shared__ bf16 Ks[2][32 * KROW];   // 9.2 KB
    __shared__ bf16 Vs[2][64 * VROW];   // 10.2 KB
    __shared__ bf16 Ps[8][16 * 32];     // 8 KB (per-wave P restripe)

    const int tid  = threadIdx.x;
    const int lane = tid & 31;
    const int w    = tid >> 5;
    const int bh   = blockIdx.x >> 4;   // 0..31
    const int qt   = (blockIdx.x & 15) * 8 + w;   // 0..127
    const int half = lane >> 4;
    const int l16  = lane & 15;
    bf16* P = Ps[w];

    const int a0 = half * 16, a1 = half * 16 + 8;

    const int krw = tid >> 3, ksg = (tid & 7) * 8;
    const int vrw = tid >> 2, vsg = (tid & 3) * 8;
    const bf16* ksrc = Kb  + ((size_t)bh * NS  + krw) * NDK + ksg;
    const bf16* vsrc = Vtb + ((size_t)bh * NDK + vrw) * NS  + vsg;

    const bf16* qrow = Qb + ((size_t)bh * NS + qt * 16 + l16) * NDK;
    v16bf qa0 = load16(qrow + a0, qrow + a1);
    v16bf qa1 = load16(qrow + 32 + a0, qrow + 32 + a1);

    async_ld_b128(&Ks[0][krw * KROW + ksg], ksrc);
    async_ld_b128(&Vs[0][vrw * VROW + vsg], vsrc);

    v8f o0 = {}, o1 = {}, o2 = {}, o3 = {};
    float mrow[8], lrow[8];
#pragma unroll
    for (int r = 0; r < 8; ++r) { mrow[r] = -1e30f; lrow[r] = 0.0f; }

    for (int t = 0; t < NS / 32; ++t) {
        if (t + 1 < NS / 32) {
            const int nb = (t + 1) & 1;
            async_ld_b128(&Ks[nb][krw * KROW + ksg],
                          ksrc + (size_t)(t + 1) * 32 * NDK);
            async_ld_b128(&Vs[nb][vrw * VROW + vsg], vsrc + (t + 1) * 32);
            asm volatile("s_wait_asynccnt 0x2" ::: "memory");
        } else {
            asm volatile("s_wait_asynccnt 0x0" ::: "memory");
        }
        __syncthreads();   // K/V block t in LDS for all waves

        const int buf = t & 1;
        const bf16* k0 = &Ks[buf][l16 * KROW + half * 16];
        const bf16* k1 = &Ks[buf][(16 + l16) * KROW + half * 16];
        v16bf kb0 = load16(k0, k0 + 8);
        v16bf kb1 = load16(k0 + 32, k0 + 40);
        v16bf kb2 = load16(k1, k1 + 8);
        v16bf kb3 = load16(k1 + 32, k1 + 40);
        v8f s0 = {}, s1 = {};
        s0 = wmma_bf16(qa0, kb0, s0);
        s0 = wmma_bf16(qa1, kb1, s0);
        s1 = wmma_bf16(qa0, kb2, s1);
        s1 = wmma_bf16(qa1, kb3, s1);

        // V tiles: issue LDS reads now; softmax below hides the DS latency
        const bf16* v0 = &Vs[buf][(0  + l16) * VROW + half * 16];
        const bf16* v1 = &Vs[buf][(16 + l16) * VROW + half * 16];
        const bf16* v2 = &Vs[buf][(32 + l16) * VROW + half * 16];
        const bf16* v3 = &Vs[buf][(48 + l16) * VROW + half * 16];
        v16bf vb0 = load16(v0, v0 + 8);
        v16bf vb1 = load16(v1, v1 + 8);
        v16bf vb2 = load16(v2, v2 + 8);
        v16bf vb3 = load16(v3, v3 + 8);

        // shape the front of the iteration: K reads, 2 WMMA, V reads, 2 WMMA
        SGB(0x100, 8);    // kb0..kb3
        SGB(0x008, 2);    // first two score WMMAs
        SGB(0x100, 8);    // vb0..vb3
        SGB(0x008, 2);    // remaining score WMMAs

        // online softmax (rows striped across the two 16-lane groups)
#pragma unroll
        for (int r = 0; r < 8; ++r) {
            float sa = s0[r] * 0.125f;           // 1/sqrt(64)
            float sb = s1[r] * 0.125f;
            float mx = rmax16(fmaxf(sa, sb));
            float mnew = fmaxf(mrow[r], mx);
            float alpha = __expf(mrow[r] - mnew);
            float p0 = __expf(sa - mnew);
            float p1 = __expf(sb - mnew);
            lrow[r] = lrow[r] * alpha + rsum16(p0 + p1);
            mrow[r] = mnew;
            o0[r] *= alpha; o1[r] *= alpha; o2[r] *= alpha; o3[r] *= alpha;
            const int row = r + half * 8;
            P[row * 32 + l16]      = (bf16)p0;
            P[row * 32 + 16 + l16] = (bf16)p1;
        }

        // C-layout -> A-layout restripe (same-wave DS ops are in order)
        v16bf pa = load16(P + l16 * 32 + a0, P + l16 * 32 + a1);

        o0 = wmma_bf16(pa, vb0, o0);
        o1 = wmma_bf16(pa, vb1, o1);
        o2 = wmma_bf16(pa, vb2, o2);
        o3 = wmma_bf16(pa, vb3, o3);

        __syncthreads();   // done reading buf before it is restaged
    }

    const int b = bh >> 4, h = bh & 15;
#pragma unroll
    for (int r = 0; r < 8; ++r) {
        const float inv = 1.0f / lrow[r];
        const int srow = qt * 16 + r + half * 8;
        bf16* dst = Ab + ((size_t)(b * NS + srow)) * ND + h * NDK;
        dst[0 * 16 + l16] = (bf16)(o0[r] * inv);
        dst[1 * 16 + l16] = (bf16)(o1[r] * inv);
        dst[2 * 16 + l16] = (bf16)(o2[r] * inv);
        dst[3 * 16 + l16] = (bf16)(o3[r] * inv);
    }
}

// ---------------------------------------------------------------------------
// Kernel 4: output projection  out = attn @ Wo^T (fp32 store).
// ---------------------------------------------------------------------------
__global__ __launch_bounds__(256) void out_gemm(
    const bf16* __restrict__ Ab, const bf16* __restrict__ wo,
    float* __restrict__ out) {

    __shared__ bf16 Bs[2][64 * BROW];

    const int tid  = threadIdx.x;
    const int lane = tid & 31;
    const int w    = tid >> 5;
    const int colBlock = blockIdx.x & 15;
    const int rowGroup = blockIdx.x >> 4;
    const int rowTile  = rowGroup * 8 + w;
    const int half = lane >> 4;
    const int l16  = lane & 15;

    const int srow = tid >> 2;
    const int scol = (tid & 3) * 8;
    const bf16* wsrc = wo + (size_t)(colBlock * 64 + srow) * ND + scol;

    const bf16* arow = Ab + (size_t)(rowTile * 16 + l16) * ND;
    const int a0 = half * 16, a1 = half * 16 + 8;

    v8f acc0 = {}, acc1 = {}, acc2 = {}, acc3 = {};

    async_ld_b128(&Bs[0][srow * BROW + scol], wsrc);
    v16bf a_c = load16(arow + a0, arow + a1);

    for (int kk = 0; kk < ND / 32; ++kk) {
        if (kk + 1 < ND / 32) {
            async_ld_b128(&Bs[(kk + 1) & 1][srow * BROW + scol],
                          wsrc + (kk + 1) * 32);
            asm volatile("s_wait_asynccnt 0x1" ::: "memory");
        } else {
            asm volatile("s_wait_asynccnt 0x0" ::: "memory");
        }
        __syncthreads();

        const int kn = (kk + 1 < ND / 32 ? kk + 1 : kk) * 32;
        v16bf a_n = load16(arow + kn + a0, arow + kn + a1);

        const bf16* Bp = &Bs[kk & 1][0];
        const bf16* p0 = Bp + (0  + l16) * BROW + half * 16;
        const bf16* p1 = Bp + (16 + l16) * BROW + half * 16;
        const bf16* p2 = Bp + (32 + l16) * BROW + half * 16;
        const bf16* p3 = Bp + (48 + l16) * BROW + half * 16;

        v16bf b0 = load16(p0, p0 + 8);
        v16bf b1 = load16(p1, p1 + 8);
        v16bf b2 = load16(p2, p2 + 8);
        v16bf b3 = load16(p3, p3 + 8);
        acc0 = wmma_bf16(a_c, b0, acc0);
        acc1 = wmma_bf16(a_c, b1, acc1);
        acc2 = wmma_bf16(a_c, b2, acc2);
        acc3 = wmma_bf16(a_c, b3, acc3);

        // shape: 2 tiles ahead of each WMMA
        SGB(0x100, 4);    // tiles 0,1
        SGB(0x008, 1);    // wmma 0
        SGB(0x100, 2);    // tile 2
        SGB(0x008, 1);    // wmma 1
        SGB(0x100, 2);    // tile 3
        SGB(0x008, 2);    // wmma 2,3

        a_c = a_n;
        __syncthreads();
    }

#pragma unroll
    for (int r = 0; r < 8; ++r) {
        const int m = rowTile * 16 + r + half * 8;
        float* orow = out + (size_t)m * ND + colBlock * 64 + l16;
        orow[0]  = acc0[r];
        orow[16] = acc1[r];
        orow[32] = acc2[r];
        orow[48] = acc3[r];
    }
}

// ---------------------------------------------------------------------------
extern "C" void kernel_launch(void* const* d_in, const int* in_sizes, int n_in,
                              void* d_out, int out_size, void* d_ws, size_t ws_size,
                              hipStream_t stream) {
    const float* x  = (const float*)d_in[0];
    const float* Wq = (const float*)d_in[1];
    const float* Wk = (const float*)d_in[2];
    const float* Wv = (const float*)d_in[3];
    const float* Wo = (const float*)d_in[4];

    const size_t NX = (size_t)NM * ND;
    const size_t NW = (size_t)ND * ND;

    bf16* xb  = (bf16*)d_ws;
    bf16* wqb = xb  + NX;
    bf16* wkb = wqb + NW;
    bf16* wvb = wkb + NW;
    bf16* wob = wvb + NW;
    bf16* Qb  = wob + NW;
    bf16* Kb  = Qb  + NX;
    bf16* Vtb = Kb  + NX;
    bf16* Ab  = Vtb + NX;   // ~48 MB total workspace

    cvt_bf16<<<(int)(NX / 1024), 256, 0, stream>>>(x,  xb,  (int)NX);
    cvt_bf16<<<(int)(NW / 1024), 256, 0, stream>>>(Wq, wqb, (int)NW);
    cvt_bf16<<<(int)(NW / 1024), 256, 0, stream>>>(Wk, wkb, (int)NW);
    cvt_bf16<<<(int)(NW / 1024), 256, 0, stream>>>(Wv, wvb, (int)NW);
    cvt_bf16<<<(int)(NW / 1024), 256, 0, stream>>>(Wo, wob, (int)NW);

    qkv_gemm<<<512, 256, 0, stream>>>(xb, wqb, wkb, wvb, Qb, Kb, Vtb);
    flash_attn<<<512, 256, 0, stream>>>(Qb, Kb, Vtb, Ab);
    out_gemm<<<512, 256, 0, stream>>>(Ab, wob, (float*)d_out);
}
